// LSTM_81157702025332
// MI455X (gfx1250) — matrix operand order repaired and checked by reading, not compiled, not run on hardware
//
#include <hip/hip_runtime.h>

#define OBS_LEN     9
#define NUM_TRACKS  20000
#define EMB_DIM     64
#define HID         128
#define GATES       (4 * HID)                            // 512
#define KROW        (EMB_DIM + HID)                      // 192 halfs per packed row
#define N_PREDICT   12
#define NSTEPS      (OBS_LEN - 1 + 1 + (N_PREDICT - 2))  // 19
#define SCALE_F     4.0f

#define TILE_TRACKS     16
#define WAVES_PER_BLOCK 16
#define BLOCK_THREADS   (WAVES_PER_BLOCK * 32)           // 512

// block strides, in halfs
#define G_STRIDE  (16 * KROW)          // 3072 halfs  = 6144 B  per gate block
#define T_STRIDE  (4 * G_STRIDE)       // 12288 halfs = 24576 B per hid-tile

typedef _Float16     v8h  __attribute__((ext_vector_type(8)));
typedef _Float16     v16h __attribute__((ext_vector_type(16)));
typedef float        v4f  __attribute__((ext_vector_type(4)));
typedef float        v8f  __attribute__((ext_vector_type(8)));
typedef unsigned int v4u  __attribute__((ext_vector_type(4)));

union HPack { v8h h; v4u u; };

// ---------------- LDS layout (bytes) ----------------
#define OFF_WBLK   0                                  // 512*192*2 = 196608
#define OFF_BIAS   (OFF_WBLK + GATES * KROW * 2)      // 196608 ([t][g][16] f32)
#define OFF_WEMB4  (OFF_BIAS + GATES * 4)             // 198656 (64 x {wx,wy,wz,b})
#define OFF_WNA    (OFF_WEMB4 + EMB_DIM * 4 * 4)      // 199680 (W_n as 16x128 f16 A)
#define OFF_BN     (OFF_WNA + 16 * HID * 2)           // 203776
#define SMEM_BYTES (OFF_BN + 16)                      // 203792 < 320KB/WGP

#if defined(__has_builtin)
#if __has_builtin(__builtin_amdgcn_tanhf)
#define HAVE_HW_TANH 1
#endif
#endif

__device__ __forceinline__ float tanh_fast(float x) {
#ifdef HAVE_HW_TANH
  return __builtin_amdgcn_tanhf(x);                       // v_tanh_f32
#else
  return 1.0f - 2.0f * __builtin_amdgcn_rcpf(1.0f + __expf(2.0f * x));
#endif
}
__device__ __forceinline__ float sigmoid_fast(float x) {
#ifdef HAVE_HW_TANH
  return fmaf(0.5f, __builtin_amdgcn_tanhf(0.5f * x), 0.5f);
#else
  return __builtin_amdgcn_rcpf(1.0f + __expf(-x));        // v_exp + v_rcp
#endif
}

// Two 16B LDS loads -> one 16x32 f16 A-fragment (contiguous k order: K = 16*halfsel + 0..15).
__device__ __forceinline__ v16h frag_at(const _Float16* p) {
  v8h lo = *(const v8h*)(p);        // K = kbase+koff+0..7   (ds_load_b128)
  v8h hi = *(const v8h*)(p + 8);    // K = kbase+koff+8..15  (ds_load_b128)
  return __builtin_shufflevector(lo, hi, 0,1,2,3,4,5,6,7,8,9,10,11,12,13,14,15);
}

// One 16(gate-rows) x 16(tracks) tile of gates^T for hid-tile t, gate g.
// tp      = Wblk + t*T_STRIDE + row*KROW + koff   (per-lane, per-t base)
// biaslane= bias_l + halfsel*8                     (per-lane base; rest immediate)
__device__ __forceinline__ v8f gate_tile(const _Float16* tp, const float* biaslane,
                                         int t, int g,
                                         const v16h* embB, const v16h* hB) {
  const v4f* bp = (const v4f*)(biaslane + (t * 64 + g * 16));  // 32B-aligned
  v4f b0 = bp[0], b1 = bp[1];
  v8f acc;
  #pragma unroll
  for (int r = 0; r < 4; ++r) { acc[r] = b0[r]; acc[r + 4] = b1[r]; }
  const _Float16* gp = tp + g * G_STRIDE;
  #pragma unroll
  for (int s = 0; s < 2; ++s) {  // K = 64 over embedding (k offsets 0,32)
    v16h a = frag_at(gp + s * 32);
    acc = __builtin_amdgcn_wmma_f32_16x16x32_f16(false, a, false, embB[s],
                                                 (short)0, acc, false, false);
  }
  #pragma unroll
  for (int s = 0; s < 4; ++s) {  // K = 128 over hidden (k offsets 64..)
    v16h a = frag_at(gp + EMB_DIM + s * 32);
    acc = __builtin_amdgcn_wmma_f32_16x16x32_f16(false, a, false, hB[s],
                                                 (short)0, acc, false, false);
  }
  return acc;
}

// Stage LSTM weights into per-(t,g) blocks: Wblk[((t*4+g)*16+row)*192 + k],
// k<64 -> W_ih row, k>=64 -> W_hh row.  gate row gr = g*128 + t*16 + row.
__device__ __forceinline__ void stage_lstm_weights(const float* Wih_g, const float* Whh_g,
                                                   const float* bih_g, const float* bhh_g,
                                                   _Float16* Wblk, float* bias_l,
                                                   int tid, int nthr) {
  for (int i = tid; i < GATES * KROW; i += nthr) {
    const int gr  = i / KROW;
    const int k   = i - gr * KROW;
    const int g   = gr >> 7;
    const int t   = (gr >> 4) & 7;
    const int row = gr & 15;
    const float v = (k < EMB_DIM) ? Wih_g[gr * EMB_DIM + k]
                                  : Whh_g[gr * HID + (k - EMB_DIM)];
    Wblk[((t * 4 + g) * 16 + row) * KROW + k] = (_Float16)v;
  }
  for (int gr = tid; gr < GATES; gr += nthr) {
    const int g   = gr >> 7;
    const int t   = (gr >> 4) & 7;
    const int row = gr & 15;
    bias_l[(t * 4 + g) * 16 + row] = bih_g[gr] + bhh_g[gr];
  }
}

__global__ __launch_bounds__(BLOCK_THREADS, 1)
void lstm_wmma_kernel(const float* __restrict__ observed,      // (9, NT, 3)
                      const float* __restrict__ W_emb,         // (64, 3)
                      const float* __restrict__ b_emb,         // (64)
                      const float* __restrict__ eWih, const float* __restrict__ eWhh,
                      const float* __restrict__ ebih, const float* __restrict__ ebhh,
                      const float* __restrict__ dWih, const float* __restrict__ dWhh,
                      const float* __restrict__ dbih, const float* __restrict__ dbhh,
                      const float* __restrict__ W_n,           // (3, 128)
                      const float* __restrict__ b_n,           // (3)
                      float* __restrict__ out)                 // rel(19,NT,3) ++ pos(19,NT,3)
{
  extern __shared__ char smem[];
  _Float16* Wblk   = (_Float16*)(smem + OFF_WBLK);
  float*    bias_l = (float*)(smem + OFF_BIAS);
  float*    Wemb4  = (float*)(smem + OFF_WEMB4);   // [64][4] = {wx,wy,wz,b_emb}
  _Float16* WnA    = (_Float16*)(smem + OFF_WNA);  // 16x128 f16, rows 3..15 zero
  float*    bn_l   = (float*)(smem + OFF_BN);

  const int tid  = threadIdx.x;
  const int nthr = blockDim.x;

  // Stage common tables + encoder LSTM weights (f32->f16).
  for (int k = tid; k < EMB_DIM; k += nthr) {
    Wemb4[k * 4 + 0] = W_emb[k * 3 + 0];
    Wemb4[k * 4 + 1] = W_emb[k * 3 + 1];
    Wemb4[k * 4 + 2] = W_emb[k * 3 + 2];
    Wemb4[k * 4 + 3] = b_emb[k];
  }
  for (int i = tid; i < 16 * HID; i += nthr) {
    const int row = i >> 7, k = i & (HID - 1);
    WnA[i] = (row < 3) ? (_Float16)W_n[row * HID + k] : (_Float16)0.0f;
  }
  if (tid < 3) bn_l[tid] = b_n[tid];
  stage_lstm_weights(eWih, eWhh, ebih, ebhh, Wblk, bias_l, tid, nthr);
  __syncthreads();

  const int  lane    = tid & 31;
  const int  halfsel = lane >> 4;                         // 0 or 1
  const int  tile    = blockIdx.x * WAVES_PER_BLOCK + (tid >> 5);
  const int  track   = tile * TILE_TRACKS + (lane & 15);  // this lane's track
  const bool valid   = track < NUM_TRACKS;
  const float NANV   = __builtin_nanf("");

  // Per-lane bases; every other displacement is a DS immediate offset.
  const _Float16* wlane    = Wblk + (lane & 15) * KROW + (halfsel << 4);
  const float*    biaslane = bias_l + halfsel * 8;
  const v4f*      embp     = (const v4f*)Wemb4 + halfsel * 16;        // + s*32 + i
  const _Float16* wnlane   = WnA + (lane & 15) * HID + (halfsel << 4); // + s*32

  // Per-lane persistent state (one track per lane; both half-waves mirror it).
  v8f  c_reg[8];   // c: 8 hid-tiles x 8 rows (f32, D layout)
  v16h hB[4];      // h^T as 4 B-fragments (f16, K=32 each)
  #pragma unroll
  for (int s = 0; s < 8; ++s)
    #pragma unroll
    for (int r = 0; r < 8; ++r) c_reg[s][r] = 0.0f;
  #pragma unroll
  for (int s = 0; s < 4; ++s)
    #pragma unroll
    for (int i = 0; i < 16; ++i) hB[s][i] = (_Float16)0.0f;

  float pm2x = NANV, pm2y = NANV, pm2z = NANV;   // pos[t-2]
  float pm1x = NANV, pm1y = NANV, pm1z = NANV;   // pos[t-1]

  #pragma unroll 1
  for (int step = 0; step < NSTEPS; ++step) {
    if (step == OBS_LEN - 1) {  // encoder done -> swap in decoder weights
      __syncthreads();
      stage_lstm_weights(dWih, dWhh, dbih, dbhh, Wblk, bias_l, tid, nthr);
      __syncthreads();
    }

    // ---- select step inputs (o1, o2) ----
    float o1x, o1y, o2x, o2y, o2z;
    if (step < OBS_LEN - 1) {                 // encoder: observed pairs
      if (valid) {
        const float* p1 = observed + ((long)step * NUM_TRACKS + track) * 3;
        const float* p2 = p1 + (long)NUM_TRACKS * 3;
        o1x = p1[0]; o1y = p1[1];
        o2x = p2[0]; o2y = p2[1]; o2z = p2[2];
      } else { o1x = o1y = o2x = o2y = o2z = NANV; }
    } else if (step == OBS_LEN - 1) {         // first decoder step
      if (track == 0) { o1x = pm2x; o1y = pm2y; }
      else if (valid) {
        const float* p1 = observed + ((long)(OBS_LEN - 1) * NUM_TRACKS + track) * 3;
        o1x = p1[0]; o1y = p1[1];
      } else { o1x = o1y = NANV; }
      o2x = pm1x; o2y = pm1y; o2z = pm1z;
    } else {                                  // autoregressive decoder
      o1x = pm2x; o1y = pm2y;
      o2x = pm1x; o2y = pm1y; o2z = pm1z;
    }

    const bool mask = !(__builtin_isnan(o1x) || __builtin_isnan(o2x));
    const float sx = (mask ? (o2x - o1x) : 0.0f) * SCALE_F;  // obs1.z treated as 0
    const float sy = (mask ? (o2y - o1y) : 0.0f) * SCALE_F;
    const float sz = (mask ? o2z : 0.0f) * SCALE_F;

    // ---- emb^T B-fragments: one ds_load_b128 per emb row, immediate offsets ----
    v16h embB[2];
    #pragma unroll
    for (int s = 0; s < 2; ++s) {
      #pragma unroll
      for (int i = 0; i < 16; ++i) {
        v4f w = embp[s * 32 + i];
        float e = fmaf(w[0], sx, fmaf(w[1], sy, fmaf(w[2], sz, w[3])));
        embB[s][i] = (_Float16)fmaxf(e, 0.0f);
      }
    }

    // ---- gates^T via WMMA, element-wise LSTM, rebuild h B-fragments ----
    v16h hBn[4];
    #pragma unroll
    for (int p4 = 0; p4 < 4; ++p4) {
      HPack pk[2];                               // packed f16 h2 for the two tiles
      #pragma unroll
      for (int hh = 0; hh < 2; ++hh) {
        const int t = 2 * p4 + hh;
        const _Float16* tp = wlane + t * T_STRIDE;   // one base covers all 4 gates
        // g then i: combine immediately so only one extra tile stays live
        v8f gg = gate_tile(tp, biaslane, t, 2, embB, hB);
        v8f gi = gate_tile(tp, biaslane, t, 0, embB, hB);
        v8f act;
        #pragma unroll
        for (int r = 0; r < 8; ++r)
          act[r] = sigmoid_fast(gi[r]) * tanh_fast(gg[r]);
        v8f gf = gate_tile(tp, biaslane, t, 1, embB, hB);
        #pragma unroll
        for (int r = 0; r < 8; ++r) {
          float c2 = fmaf(sigmoid_fast(gf[r]), c_reg[t][r], act[r]);
          c_reg[t][r] = mask ? c2 : c_reg[t][r];
          act[r] = tanh_fast(c2);                // reuse as tanh(c2)
        }
        v8f go = gate_tile(tp, biaslane, t, 3, embB, hB);
        #pragma unroll
        for (int r = 0; r < 8; ++r) {
          float hv = sigmoid_fast(go[r]) * act[r];
          pk[hh].h[r] = (_Float16)hv;            // f32 h2 dies here
        }
      }
      // D layout -> B layout for next step's h^T: packed 32-bit half-wave swap
      v4u olo, ohi;
      #pragma unroll
      for (int j = 0; j < 4; ++j) {
        olo[j] = __shfl_xor(pk[0].u[j], 16, 32);
        ohi[j] = __shfl_xor(pk[1].u[j], 16, 32);
      }
      HPack b0, b1;
      b0.u = halfsel ? ohi : pk[0].u;   // fragment halves K = 0..7 (+16*halfsel)
      b1.u = halfsel ? pk[1].u : olo;   // fragment halves K = 8..15 (+16*halfsel)
      v16h bf = __builtin_shufflevector(b0.h, b1.h,
                                        0,1,2,3,4,5,6,7,8,9,10,11,12,13,14,15);
      hBn[p4] = mask ? bf : hB[p4];     // masked tracks keep old h
    }
    #pragma unroll
    for (int s = 0; s < 4; ++s) hB[s] = hBn[s];

    // ---- output head via WMMA: normal^T = W_n(16x128, rows 3..15 = 0) @ h^T ----
    // Using masked h is safe: when mask==0 the result is replaced with NaN below.
    v8f nacc;
    #pragma unroll
    for (int r = 0; r < 8; ++r) nacc[r] = 0.0f;
    #pragma unroll
    for (int s = 0; s < 4; ++s) {
      v16h a = frag_at(wnlane + s * 32);
      nacc = __builtin_amdgcn_wmma_f32_16x16x32_f16(false, a, false, hB[s],
                                                    (short)0, nacc, false, false);
    }
    // Rows 0..2 live in the halfsel==0 half; other half holds zero rows, so a
    // single xor-16 add both combines and broadcasts to the mirror half.
    const float q0 = nacc[0] + __shfl_xor(nacc[0], 16, 32);
    const float q1 = nacc[1] + __shfl_xor(nacc[1], 16, 32);
    const float q2 = nacc[2] + __shfl_xor(nacc[2], 16, 32);
    const float n0 = mask ? (q0 + bn_l[0]) : NANV;
    const float n1 = mask ? (q1 + bn_l[1]) : NANV;
    const float n2 = mask ? (q2 + bn_l[2]) : NANV;
    const float px = o2x + n0;
    const float py = o2y + n1;
    const float pz = n2;                 // obs2.z set to 0 before add

    pm2x = pm1x; pm2y = pm1y; pm2z = pm1z;
    pm1x = px;   pm1y = py;   pm1z = pz;

    if (halfsel == 0 && valid) {
      const long base  = ((long)step * NUM_TRACKS + track) * 3;
      const long pbase = (long)NSTEPS * NUM_TRACKS * 3 + base;
      out[base + 0]  = n0; out[base + 1]  = n1; out[base + 2]  = n2;
      out[pbase + 0] = px; out[pbase + 1] = py; out[pbase + 2] = pz;
    }
  }
}

extern "C" void kernel_launch(void* const* d_in, const int* in_sizes, int n_in,
                              void* d_out, int out_size, void* d_ws, size_t ws_size,
                              hipStream_t stream) {
  (void)in_sizes; (void)n_in; (void)out_size; (void)d_ws; (void)ws_size;
  const float* observed = (const float*)d_in[0];
  // d_in[1] batch_split = [0, NT] (primary track 0), d_in[2] n_predict = 12: baked in.
  const float* W_emb = (const float*)d_in[3];
  const float* b_emb = (const float*)d_in[4];
  const float* eWih  = (const float*)d_in[5];
  const float* eWhh  = (const float*)d_in[6];
  const float* ebih  = (const float*)d_in[7];
  const float* ebhh  = (const float*)d_in[8];
  const float* dWih  = (const float*)d_in[9];
  const float* dWhh  = (const float*)d_in[10];
  const float* dbih  = (const float*)d_in[11];
  const float* dbhh  = (const float*)d_in[12];
  const float* W_n   = (const float*)d_in[13];
  const float* b_n   = (const float*)d_in[14];
  float* out = (float*)d_out;

  hipFuncSetAttribute((const void*)lstm_wmma_kernel,
                      hipFuncAttributeMaxDynamicSharedMemorySize, SMEM_BYTES);

  const int tiles  = (NUM_TRACKS + TILE_TRACKS - 1) / TILE_TRACKS;        // 1250
  const int blocks = (tiles + WAVES_PER_BLOCK - 1) / WAVES_PER_BLOCK;     // 79
  lstm_wmma_kernel<<<blocks, BLOCK_THREADS, SMEM_BYTES, stream>>>(
      observed, W_emb, b_emb, eWih, eWhh, ebih, ebhh,
      dWih, dWhh, dbih, dbhh, W_n, b_n, out);
}
